// GPTActor_74749610820279
// MI455X (gfx1250) — matrix operand order repaired and checked by reading, not compiled, not run on hardware
//
#include <hip/hip_runtime.h>
#include <hip/hip_bf16.h>

// ---------------------------------------------------------------------------
// GPT forward for MI455X (gfx1250, wave32). All dense GEMMs run through
// v_wmma_f32_16x16x32_f16 (f16 inputs, f32 accum), double-buffered LDS with
// GLOBAL_LOAD_ASYNC_TO_LDS (ASYNCcnt) when available. Weights fp32->f16 once
// per launch. Attention = wave-per-query online softmax (L2 resident).
// ---------------------------------------------------------------------------

typedef _Float16 v16h __attribute__((ext_vector_type(16)));
typedef _Float16 v8h  __attribute__((ext_vector_type(8)));
typedef float    v8f  __attribute__((ext_vector_type(8)));
typedef int      i4v  __attribute__((ext_vector_type(4)));

union FragAB { v16h v; v8h c[2]; };

#define BM 128
#define BN 128
#define BK 32
#define LDSR 48                 // halves per LDS row (96B: 16B-aligned, low conflict)
#define STG  ((BM + BN) * LDSR) // halves per stage (A rows then W rows)

#if defined(__gfx1250__) && __has_builtin(__builtin_amdgcn_global_load_async_to_lds_b128)
#define USE_ASYNC 1
#else
#define USE_ASYNC 0
#endif

typedef __attribute__((address_space(1))) i4v g_i4;   // global int4*
typedef __attribute__((address_space(3))) i4v l_i4;   // LDS    int4*

__device__ __forceinline__ void stage16(const _Float16* g, _Float16* l)
{
#if USE_ASYNC
    __builtin_amdgcn_global_load_async_to_lds_b128(
        (g_i4*)(unsigned long long)g,
        (l_i4*)(unsigned int)(unsigned long long)l, 0, 0);
#else
    *(v8h*)l = *(const v8h*)g;
#endif
}

__device__ __forceinline__ void wait_async()
{
#if USE_ASYNC
#if __has_builtin(__builtin_amdgcn_s_wait_asynccnt)
    __builtin_amdgcn_s_wait_asynccnt(0);
#else
    asm volatile("s_wait_asynccnt 0x0" ::: "memory");
#endif
#endif
}

// ---------------------------------------------------------------------------
// Tiled WMMA GEMM:  C(M,N) = A(M,K) @ W(N,K)^T + bias  [+ residual | gelu->f16]
// A, W f16 row-major (K-major). mode 0: f32 out (+optional residual add).
// mode 1: gelu(acc+bias) -> f16. Block 256 thr = 8 waves (2x4); each wave owns
// a 64x32 C region = 8 WMMA tiles. Double-buffered LDS, async staged.
// ---------------------------------------------------------------------------
__global__ __launch_bounds__(256) void gemm_wmma_f16(
    const _Float16* __restrict__ A, const _Float16* __restrict__ W,
    const float* __restrict__ bias, float* __restrict__ Cf32,
    _Float16* __restrict__ Cf16, const float* __restrict__ residual,
    int M, int N, int K, int mode)
{
    __shared__ _Float16 smem[2 * STG];   // 48 KB

    const int tid  = threadIdx.x;
    const int wave = tid >> 5;
    const int lane = tid & 31;
    const int wr   = wave >> 2;          // 0..1 (M)
    const int wc   = wave & 3;           // 0..3 (N)
    const int l16  = lane & 15;
    const int hi   = lane >> 4;          // 0/1
    const long blockM = (long)blockIdx.y * BM;
    const long blockN = (long)blockIdx.x * BN;

    v8f acc[4][2] = {};

    // stage one BK-slice (A tile 128x32, W tile 128x32) into LDS stage st
    auto stage = [&](int st, int kb) {
        _Float16* ab = smem + st * STG;
        _Float16* wb = ab + BM * LDSR;
#pragma unroll
        for (int c = 0; c < 2; ++c) {                 // A: 512 16B-chunks
            int id = tid + c * 256;
            int row = id >> 2, ck = id & 3;
            stage16(A + (blockM + row) * (long)K + kb + ck * 8,
                    ab + row * LDSR + ck * 8);
        }
#pragma unroll
        for (int c = 0; c < 2; ++c) {                 // W: 512 16B-chunks
            int id = tid + c * 256;
            int row = id >> 2, ck = id & 3;
            long n = blockN + row;
            if (n >= N) n = N - 1;                    // branchless tail clamp
            stage16(W + n * (long)K + kb + ck * 8,
                    wb + row * LDSR + ck * 8);
        }
    };

    const int nk = K / BK;
    int cur = 0;
    stage(0, 0);
    wait_async();
    __syncthreads();

    for (int kb = 0; kb < nk; ++kb) {
        int nxt = cur ^ 1;
        if (kb + 1 < nk) stage(nxt, (kb + 1) * BK);   // prefetch next slice

        const _Float16* ab = smem + cur * STG;
        const _Float16* wb = ab + BM * LDSR;

        FragAB af[4], bf[2];
#pragma unroll
        for (int i = 0; i < 4; ++i) {                 // A frags: rows wr*64 + i*16
            int arow = wr * 64 + i * 16 + l16;
            af[i].c[0] = *(const v8h*)&ab[arow * LDSR + hi * 8];       // K 0..7 / 8..15
            af[i].c[1] = *(const v8h*)&ab[arow * LDSR + hi * 8 + 16];  // K 16..23 / 24..31
        }
#pragma unroll
        for (int j = 0; j < 2; ++j) {                 // B frags: cols wc*32 + j*16
            int wrow = wc * 32 + j * 16 + l16;
            bf[j].c[0] = *(const v8h*)&wb[wrow * LDSR + hi * 8];       // K 0..7 / 8..15
            bf[j].c[1] = *(const v8h*)&wb[wrow * LDSR + hi * 8 + 16];  // K 16..23 / 24..31
        }
#pragma unroll
        for (int i = 0; i < 4; ++i)
#pragma unroll
            for (int j = 0; j < 2; ++j)
                acc[i][j] = __builtin_amdgcn_wmma_f32_16x16x32_f16(
                    false, af[i].v, false, bf[j].v,
                    (short)0, acc[i][j], false, false);

        wait_async();          // next-stage LDS writes landed (mine)
        __syncthreads();       // everyone done reading cur / writing nxt
        cur = nxt;
    }

    // --- epilogue: bias / residual / gelu
#pragma unroll
    for (int i = 0; i < 4; ++i)
#pragma unroll
        for (int j = 0; j < 2; ++j) {
            long n = blockN + wc * 32 + j * 16 + l16;
            if (n >= N) continue;
            float bv = bias ? bias[n] : 0.0f;
#pragma unroll
            for (int r = 0; r < 8; ++r) {
                long m   = blockM + wr * 64 + i * 16 + hi * 8 + r;
                long idx = m * (long)N + n;
                float v  = acc[i][j][r] + bv;
                if (mode == 0) {
                    if (residual) v += residual[idx];
                    Cf32[idx] = v;
                } else {
                    float t = 0.7978845608028654f * (v + 0.044715f * v * v * v);
                    Cf16[idx] = (_Float16)(0.5f * v * (1.0f + tanhf(t)));
                }
            }
        }
}

// ---------------------------------------------------------------------------
// Embedding: h[b,t,:] = token_emb[x[b,t]] + pos_emb[t]
// ---------------------------------------------------------------------------
__global__ void embed_kernel(const int* __restrict__ x,
                             const float* __restrict__ tok,
                             const float* __restrict__ pos,
                             float* __restrict__ h, int T, int Dd)
{
    long row = blockIdx.x;
    int  t   = (int)(row % T);
    long id  = x[row];
    for (int d = threadIdx.x; d < Dd; d += blockDim.x)
        h[row * Dd + d] = tok[id * Dd + d] + pos[(long)t * Dd + d];
}

// ---------------------------------------------------------------------------
// LayerNorm, writes f16 (GEMM A operand)
// ---------------------------------------------------------------------------
__global__ __launch_bounds__(256) void ln_kernel(
    const float* __restrict__ xin, const float* __restrict__ g,
    const float* __restrict__ bta, _Float16* __restrict__ out, int Dd)
{
    long row = blockIdx.x;
    const float* xr = xin + row * Dd;
    __shared__ float rs[256], rq[256];
    float s = 0.f, q = 0.f;
    for (int d = threadIdx.x; d < Dd; d += 256) { float v = xr[d]; s += v; q += v * v; }
    rs[threadIdx.x] = s; rq[threadIdx.x] = q;
    __syncthreads();
    for (int off = 128; off > 0; off >>= 1) {
        if (threadIdx.x < off) { rs[threadIdx.x] += rs[threadIdx.x + off];
                                 rq[threadIdx.x] += rq[threadIdx.x + off]; }
        __syncthreads();
    }
    float mu   = rs[0] / Dd;
    float var  = rq[0] / Dd - mu * mu;
    float rstd = rsqrtf(var + 1e-5f);
    for (int d = threadIdx.x; d < Dd; d += 256)
        out[row * Dd + d] = (_Float16)((xr[d] - mu) * rstd * g[d] + bta[d]);
}

// ---------------------------------------------------------------------------
// Attention: one wave per (b, h, tq). Online softmax over causal keys.
// qkv layout (B*T, 3D): q=[0,D), k=[D,2D), v=[2D,3D); head h uses dims h*64..+63.
// All shuffles run with full EXEC (addresses clamped, no divergent branches).
// ---------------------------------------------------------------------------
__global__ __launch_bounds__(256) void attn_kernel(
    const float* __restrict__ qkv, const float* __restrict__ amask,
    _Float16* __restrict__ wv16, int T, int Hh, int Dd)
{
    long gw   = (long)blockIdx.x * 8 + (threadIdx.x >> 5);
    int  lane = threadIdx.x & 31;
    int  tq   = (int)(gw % T);
    int  h    = (int)((gw / T) % Hh);
    int  b    = (int)(gw / ((long)T * Hh));

    const float* qrow  = qkv + ((long)(b * T + tq)) * 3 * Dd + (long)h * 64;
    const float* kbase = qkv + ((long)b * T) * 3 * Dd + Dd + (long)h * 64;
    const float* vbase = qkv + ((long)b * T) * 3 * Dd + 2 * Dd + (long)h * 64;

    float q0 = qrow[2 * lane], q1 = qrow[2 * lane + 1];
    float runm = -1e30f, runs = 0.f, o0 = 0.f, o1 = 0.f;

    for (int base = 0; base <= tq; base += 32) {
        int j  = base + lane;
        int jc = (j <= tq) ? j : tq;               // clamp: keep EXEC full
        const float* kr = kbase + (long)jc * 3 * Dd;
        float acc = 0.f;
#pragma unroll
        for (int d2 = 0; d2 < 32; ++d2) {
            float qa = __shfl(q0, d2), qb = __shfl(q1, d2);
            acc += qa * kr[2 * d2] + qb * kr[2 * d2 + 1];
        }
        float s = (j <= tq)
                    ? acc * 0.125f + (1.0f - amask[(long)b * T + j]) * (-1e30f)
                    : -1e30f;
        float cm = s;
#pragma unroll
        for (int off = 16; off > 0; off >>= 1) cm = fmaxf(cm, __shfl_xor(cm, off));
        float newm = fmaxf(runm, cm);
        float corr = expf(runm - newm);
        float p    = expf(s - newm);
        float ps   = p;
#pragma unroll
        for (int off = 16; off > 0; off >>= 1) ps += __shfl_xor(ps, off);
        runs = runs * corr + ps;
        o0 *= corr; o1 *= corr;
        const float* vr = vbase + (long)base * 3 * Dd;
        for (int jj = 0; jj < 32; ++jj) {
            float pj = __shfl(p, jj);
            const float* v = vr + (long)jj * 3 * Dd;
            o0 += pj * v[2 * lane];
            o1 += pj * v[2 * lane + 1];
        }
        runm = newm;
    }
    float inv = 1.0f / runs;
    long  o   = ((long)(b * T + tq)) * Dd + (long)h * 64 + 2 * lane;
    wv16[o]     = (_Float16)(o0 * inv);
    wv16[o + 1] = (_Float16)(o1 * inv);
}

// ---------------------------------------------------------------------------
// fp32 -> f16 weight conversion
// ---------------------------------------------------------------------------
__global__ void cvt_f32_f16(const float* __restrict__ in,
                            _Float16* __restrict__ out, long n)
{
    long i = (long)blockIdx.x * blockDim.x + threadIdx.x;
    if (i < n) out[i] = (_Float16)in[i];
}

// ---------------------------------------------------------------------------
// Tiny heads: diffw (5) + diffstep (3) dot products per row
// ---------------------------------------------------------------------------
__global__ __launch_bounds__(256) void heads_kernel(
    const _Float16* __restrict__ hn, const float* __restrict__ w_head,
    const float* __restrict__ step_head, float* __restrict__ out5,
    float* __restrict__ out3, int Dd)
{
    long row = blockIdx.x;
    __shared__ float red[256];
    for (int hh = 0; hh < 8; ++hh) {
        const float* w = (hh < 5) ? (w_head + (long)hh * Dd)
                                  : (step_head + (long)(hh - 5) * Dd);
        float s = 0.f;
        for (int d = threadIdx.x; d < Dd; d += 256)
            s += (float)hn[row * Dd + d] * w[d];
        red[threadIdx.x] = s;
        __syncthreads();
        for (int off = 128; off > 0; off >>= 1) {
            if (threadIdx.x < off) red[threadIdx.x] += red[threadIdx.x + off];
            __syncthreads();
        }
        if (threadIdx.x == 0) {
            if (hh < 5) out5[row * 5 + hh]       = red[0];
            else        out3[row * 3 + (hh - 5)] = red[0];
        }
        __syncthreads();
    }
}

// ---------------------------------------------------------------------------
extern "C" void kernel_launch(void* const* d_in, const int* in_sizes, int n_in,
                              void* d_out, int out_size, void* d_ws, size_t ws_size,
                              hipStream_t stream)
{
    constexpr int Bc = 4, Tc = 1024, Dc = 1024, Hc = 16, Lc = 4, FFc = 4096;
    const int M = Bc * Tc;                 // 4096 rows
    const int V = in_sizes[18] / Dc;       // 50257

    const int*   x      = (const int*)  d_in[0];
    const float* amask  = (const float*)d_in[1];
    const float* tok    = (const float*)d_in[2];
    const float* pos    = (const float*)d_in[3];
    const float* ln1_g  = (const float*)d_in[4];
    const float* ln1_b  = (const float*)d_in[5];
    const float* qkv_w  = (const float*)d_in[6];
    const float* qkv_b  = (const float*)d_in[7];
    const float* out_w  = (const float*)d_in[8];
    const float* out_b  = (const float*)d_in[9];
    const float* ln2_g  = (const float*)d_in[10];
    const float* ln2_b  = (const float*)d_in[11];
    const float* fc1_w  = (const float*)d_in[12];
    const float* fc1_b  = (const float*)d_in[13];
    const float* fc2_w  = (const float*)d_in[14];
    const float* fc2_b  = (const float*)d_in[15];
    const float* lnf_g  = (const float*)d_in[16];
    const float* lnf_b  = (const float*)d_in[17];
    const float* lm_w   = (const float*)d_in[18];
    const float* w_head = (const float*)d_in[19];
    const float* s_head = (const float*)d_in[20];
    float* out = (float*)d_out;

    // -------- workspace carve (256B aligned) --------
    char* p = (char*)d_ws;
    auto alloc = [&](size_t bytes) {
        void* r = (void*)p;
        p += (bytes + 255) & ~(size_t)255;
        return r;
    };
    const long n_qkvw = (long)Lc * 3 * Dc * Dc;
    const long n_outw = (long)Lc * Dc * Dc;
    const long n_fc1w = (long)Lc * FFc * Dc;
    const long n_fc2w = (long)Lc * Dc * FFc;
    const long n_lmw  = (long)V * Dc;

    _Float16* w16_qkv = (_Float16*)alloc(n_qkvw * 2);
    _Float16* w16_out = (_Float16*)alloc(n_outw * 2);
    _Float16* w16_fc1 = (_Float16*)alloc(n_fc1w * 2);
    _Float16* w16_fc2 = (_Float16*)alloc(n_fc2w * 2);
    _Float16* w16_lm  = (_Float16*)alloc(n_lmw  * 2);
    float*    hbuf    = (float*)   alloc((long)M * Dc * 4);
    _Float16* xn16    = (_Float16*)alloc((long)M * Dc * 2);
    float*    qkvbuf  = (float*)   alloc((long)M * 3 * Dc * 4);
    _Float16* wv16    = (_Float16*)alloc((long)M * Dc * 2);
    _Float16* ff16    = (_Float16*)alloc((long)M * FFc * 2);
    _Float16* hn16    = (_Float16*)alloc((long)M * Dc * 2);

    auto cvt = [&](const float* src, _Float16* dst, long n) {
        unsigned blocks = (unsigned)((n + 255) / 256);
        cvt_f32_f16<<<blocks, 256, 0, stream>>>(src, dst, n);
    };
    auto gemm = [&](const _Float16* A, const _Float16* W, const float* bias,
                    float* Cf, _Float16* Ch, const float* res,
                    int Mm, int Nn, int Kk, int mode) {
        dim3 g((Nn + BN - 1) / BN, Mm / BM);
        gemm_wmma_f16<<<g, 256, 0, stream>>>(A, W, bias, Cf, Ch, res,
                                             Mm, Nn, Kk, mode);
    };

    // -------- weight conversion (once per launch, deterministic) --------
    cvt(qkv_w, w16_qkv, n_qkvw);
    cvt(out_w, w16_out, n_outw);
    cvt(fc1_w, w16_fc1, n_fc1w);
    cvt(fc2_w, w16_fc2, n_fc2w);
    cvt(lm_w,  w16_lm,  n_lmw);

    // -------- embedding --------
    embed_kernel<<<M, 256, 0, stream>>>(x, tok, pos, hbuf, Tc, Dc);

    // -------- transformer layers --------
    for (int l = 0; l < Lc; ++l) {
        ln_kernel<<<M, 256, 0, stream>>>(hbuf, ln1_g + (long)l * Dc,
                                         ln1_b + (long)l * Dc, xn16, Dc);
        gemm(xn16, w16_qkv + (long)l * 3 * Dc * Dc, qkv_b + (long)l * 3 * Dc,
             qkvbuf, nullptr, nullptr, M, 3 * Dc, Dc, 0);
        attn_kernel<<<(Bc * Hc * Tc) / 8, 256, 0, stream>>>(qkvbuf, amask,
                                                            wv16, Tc, Hc, Dc);
        gemm(wv16, w16_out + (long)l * Dc * Dc, out_b + (long)l * Dc,
             hbuf, nullptr, hbuf, M, Dc, Dc, 0);          // + residual
        ln_kernel<<<M, 256, 0, stream>>>(hbuf, ln2_g + (long)l * Dc,
                                         ln2_b + (long)l * Dc, xn16, Dc);
        gemm(xn16, w16_fc1 + (long)l * FFc * Dc, fc1_b + (long)l * FFc,
             nullptr, ff16, nullptr, M, FFc, Dc, 1);      // gelu -> f16
        gemm(ff16, w16_fc2 + (long)l * Dc * FFc, fc2_b + (long)l * Dc,
             hbuf, nullptr, hbuf, M, Dc, FFc, 0);         // + residual
    }

    // -------- final LN, lm head, tiny heads --------
    ln_kernel<<<M, 256, 0, stream>>>(hbuf, lnf_g, lnf_b, hn16, Dc);
    gemm(hn16, w16_lm, nullptr, out, nullptr, nullptr, M, V, Dc, 0);
    heads_kernel<<<M, 256, 0, stream>>>(hn16, w_head, s_head,
                                        out + (long)M * V,
                                        out + (long)M * V + (long)M * 5, Dc);
}
